// LinearAttention_42356967473254
// MI455X (gfx1250) — matrix operand order repaired and checked by reading, not compiled, not run on hardware
//
#include <hip/hip_runtime.h>
#include <hip/hip_bf16.h>
#include <cstdint>
#include <cstddef>

// ---------------------------------------------------------------------------
// Linear attention for MI455X (gfx1250): bf16 WMMA (16x16x32), f32 accum,
// async global->LDS staging (ASYNCcnt) with double buffering. wave32.
// ---------------------------------------------------------------------------

typedef __bf16 bf16;
typedef __attribute__((ext_vector_type(16))) __bf16 v16bf;
typedef __attribute__((ext_vector_type(8)))  float  v8f;

#define SEQ_N   8192
#define DIM     1024
#define NHEAD   16
#define HD      64
#define BATCH   2
#define MROWS   (BATCH * SEQ_N)   // 16384
#define NCHUNK  8                 // kv reduction chunks along N
#define CHUNK   (SEQ_N / NCHUNK)  // 1024

__device__ __forceinline__ unsigned short f2bfu(float f) {
  unsigned u = __float_as_uint(f);
  u += 0x7FFFu + ((u >> 16) & 1u);           // round-to-nearest-even
  return (unsigned short)(u >> 16);
}
__device__ __forceinline__ float bfu2f(unsigned short h) {
  return __uint_as_float(((unsigned)h) << 16);
}

// Async 16-byte global -> LDS copy (per lane). LDS address = low 32 bits of
// the flat address (ISA: LDS aperture maps addr[31:0]).
__device__ __forceinline__ void async_b128(const void* gsrc, void* ldst) {
  unsigned           lds = (unsigned)(uintptr_t)ldst;
  unsigned long long ga  = (unsigned long long)(uintptr_t)gsrc;
  asm volatile("global_load_async_to_lds_b128 %0, %1, off"
               :: "v"(lds), "v"(ga) : "memory");
}
__device__ __forceinline__ void wait_async0() {
  asm volatile("s_wait_asynccnt 0x0" ::: "memory");
}

// --------------------------- elementwise convert ---------------------------
__global__ void cvt_f32_to_bf16(const float* __restrict__ src,
                                unsigned short* __restrict__ dst, int n) {
  int i = blockIdx.x * blockDim.x + threadIdx.x;
  if (i < n) dst[i] = f2bfu(src[i]);
}

// ------------------------------- main GEMM ---------------------------------
// C[i][j] = epilogue( sum_d A[i][d] * W[j][d] )   (A @ W^T)
// A: MROWS x DIM bf16 row-major, W: DIM x DIM bf16 row-major.
// MODE 0: phi (elu+1) + bf16 store; MODE 1: bf16 store; MODE 2: f32 store.
// Block tile 128x128, BK=32; 256 threads = 8 waves (4 along M x 2 along N).
// Both LDS tiles are [row][k] so every WMMA fragment is contiguous b128 reads,
// and staging is a pure row copy done with async global->LDS b128 ops,
// double-buffered across K-steps.
template <int MODE>
__global__ __launch_bounds__(256) void gemm_xwT(
    const unsigned short* __restrict__ A,
    const unsigned short* __restrict__ W,
    unsigned short* __restrict__ outB,
    float* __restrict__ outF) {
  __shared__ __align__(16) unsigned short sA[2][128 * 32];   // [m][k]
  __shared__ __align__(16) unsigned short sB[2][128 * 32];   // [n][k]

  const int t    = threadIdx.x;
  const int lane = t & 31;
  const int wid  = t >> 5;
  const int wm   = wid & 3;        // 4 waves along M (32 rows each)
  const int wn   = wid >> 2;       // 2 waves along N (64 cols each)
  const int m0   = blockIdx.y * 128;
  const int n0   = blockIdx.x * 128;
  const int g    = lane >> 4;      // lane group (0: lanes 0-15, 1: 16-31)
  const int l16  = lane & 15;

  // staging: thread t copies 16 bf16 of row (t>>1), half (t&1) for both tiles
  const int rS  = t >> 1;
  const int hS  = (t & 1) * 16;
  const unsigned short* gA = A + (size_t)(m0 + rS) * DIM + hS;
  const unsigned short* gB = W + (size_t)(n0 + rS) * DIM + hS;

  const v8f vzero = {0.f, 0.f, 0.f, 0.f, 0.f, 0.f, 0.f, 0.f};
  v8f acc[2][4];
#pragma unroll
  for (int mt = 0; mt < 2; ++mt)
#pragma unroll
    for (int nt = 0; nt < 4; ++nt) acc[mt][nt] = vzero;

  // prologue: stage k0 = 0 into buffer 0
  {
    unsigned short* dA = &sA[0][rS * 32 + hS];
    unsigned short* dB = &sB[0][rS * 32 + hS];
    async_b128(gA, dA);
    async_b128(gA + 8, dA + 8);
    async_b128(gB, dB);
    async_b128(gB + 8, dB + 8);
  }
  wait_async0();
  __syncthreads();

  int cur = 0;
  for (int k0 = 0; k0 < DIM; k0 += 32) {
    // stage next K-slab into the other buffer (overlaps with WMMA below)
    if (k0 + 32 < DIM) {
      unsigned short* dA = &sA[cur ^ 1][rS * 32 + hS];
      unsigned short* dB = &sB[cur ^ 1][rS * 32 + hS];
      async_b128(gA + (k0 + 32), dA);
      async_b128(gA + (k0 + 32) + 8, dA + 8);
      async_b128(gB + (k0 + 32), dB);
      async_b128(gB + (k0 + 32) + 8, dB + 8);
    }

    // A fragments: 16x32 bf16, lane layout => two contiguous int4 LDS reads.
    union AF { v16bf v; int4 q[2]; } af[2];
#pragma unroll
    for (int mt = 0; mt < 2; ++mt) {
      const unsigned short* p = &sA[cur][(wm * 32 + mt * 16 + l16) * 32];
      af[mt].q[0] = *(const int4*)(p + g * 16);
      af[mt].q[1] = *(const int4*)(p + g * 16 + 8);
    }
#pragma unroll
    for (int nt = 0; nt < 4; ++nt) {
      // B fragment: k = 16g + 2j pairs at fixed n -> 32 contiguous bytes.
      union BF { v16bf v; int4 q[2]; } bf;
      const unsigned short* p = &sB[cur][(wn * 64 + nt * 16 + l16) * 32 + g * 16];
      bf.q[0] = *(const int4*)(p);
      bf.q[1] = *(const int4*)(p + 8);
#pragma unroll
      for (int mt = 0; mt < 2; ++mt)
        acc[mt][nt] = __builtin_amdgcn_wmma_f32_16x16x32_bf16(
            false, af[mt].v, false, bf.v, (short)0, acc[mt][nt], false, false);
    }

    wait_async0();
    __syncthreads();
    cur ^= 1;
  }

  // Epilogue. C layout: lane L, vgpr r -> (M = r + 8*(L>>4), N = L&15).
#pragma unroll
  for (int mt = 0; mt < 2; ++mt) {
#pragma unroll
    for (int nt = 0; nt < 4; ++nt) {
      const int gn = n0 + wn * 64 + nt * 16 + l16;
#pragma unroll
      for (int r = 0; r < 8; ++r) {
        const int gm = m0 + wm * 32 + mt * 16 + r + 8 * g;
        float v = acc[mt][nt][r];
        if (MODE == 0) v = (v > 0.f) ? (v + 1.f) : __expf(v);  // elu + 1
        if (MODE == 2) {
          outF[(size_t)gm * DIM + gn] = v;
        } else {
          outB[(size_t)gm * DIM + gn] = f2bfu(v);
        }
      }
    }
  }
}

// ----------------------- kv = K^T V  (per b,h, chunked) ---------------------
// Block = (bh, chunk). 128 threads = 4 waves; wave w computes kv rows
// d = 16w..16w+15 over all 64 V columns. K and V tiles are staged TRANSPOSED
// in LDS ([d][n] / [m][n]) so both WMMA fragments are contiguous b128 reads.
// Partials go to per-chunk slots; reduced deterministically afterwards.
__global__ __launch_bounds__(128) void kv_accum(
    const unsigned short* __restrict__ Kb,
    const unsigned short* __restrict__ Vb,
    float* __restrict__ kvPart,      // [NCHUNK][32][64*64]
    float* __restrict__ ksumPart) {  // [NCHUNK][32][64]
  __shared__ __align__(16) unsigned short sKt[64 * 32];   // [d][nn]
  __shared__ __align__(16) unsigned short sVt[64 * 32];   // [m][nn]

  const int t    = threadIdx.x;
  const int lane = t & 31;
  const int wid  = t >> 5;
  const int g    = lane >> 4;
  const int l16  = lane & 15;
  const int bh   = blockIdx.x;
  const int b    = bh >> 4;
  const int h    = bh & 15;
  const int nbase = blockIdx.y * CHUNK;

  const v8f vzero = {0.f, 0.f, 0.f, 0.f, 0.f, 0.f, 0.f, 0.f};
  v8f acc[4];
#pragma unroll
  for (int mt = 0; mt < 4; ++mt) acc[mt] = vzero;
  float ks = 0.f;

  for (int ns = 0; ns < CHUNK; ns += 32) {
    // stage transposed: thread t covers row nn = t>>2, cols q*16..q*16+15
    {
      const int nn = t >> 2;
      const int q  = t & 3;
      const size_t src = (size_t)(b * SEQ_N + nbase + ns + nn) * DIM + h * HD + q * 16;
      union { int4 v[2]; unsigned short hh[16]; } kx, vx;
      kx.v[0] = ((const int4*)(Kb + src))[0];
      kx.v[1] = ((const int4*)(Kb + src))[1];
      vx.v[0] = ((const int4*)(Vb + src))[0];
      vx.v[1] = ((const int4*)(Vb + src))[1];
#pragma unroll
      for (int i = 0; i < 16; ++i) {
        sKt[(q * 16 + i) * 32 + nn] = kx.hh[i];
        sVt[(q * 16 + i) * 32 + nn] = vx.hh[i];
      }
    }
    __syncthreads();

    // k-sum partials (waves 0-1, wave-uniform branch, contiguous row)
    if (t < 64) {
#pragma unroll 8
      for (int nn = 0; nn < 32; ++nn) ks += bfu2f(sKt[t * 32 + nn]);
    }

    // A fragment = K^T tile: A[dd][nn] = sKt[d][nn] -> two contiguous int4.
    union AF { v16bf v; int4 q[2]; } af;
    {
      const unsigned short* p = &sKt[(wid * 16 + l16) * 32];
      af.q[0] = *(const int4*)(p + g * 16);
      af.q[1] = *(const int4*)(p + g * 16 + 8);
    }
#pragma unroll
    for (int mt = 0; mt < 4; ++mt) {
      union BF { v16bf v; int4 q[2]; } bf;
      const unsigned short* p = &sVt[(mt * 16 + l16) * 32 + g * 16];
      bf.q[0] = *(const int4*)(p);
      bf.q[1] = *(const int4*)(p + 8);
      acc[mt] = __builtin_amdgcn_wmma_f32_16x16x32_bf16(
          false, af.v, false, bf.v, (short)0, acc[mt], false, false);
    }
    __syncthreads();
  }

  float* kvOut = kvPart + ((size_t)blockIdx.y * 32 + bh) * (HD * HD);
#pragma unroll
  for (int mt = 0; mt < 4; ++mt)
#pragma unroll
    for (int r = 0; r < 8; ++r)
      kvOut[(wid * 16 + r + 8 * g) * HD + mt * 16 + l16] = acc[mt][r];
  if (t < 64) ksumPart[((size_t)blockIdx.y * 32 + bh) * HD + t] = ks;
}

// Reduce kv chunks; pack to bf16 pairs in [m][p] order so attn B-fragments
// are contiguous: kvpk[bh][m][p] = (kv[2p][m], kv[2p+1][m]).
__global__ void kv_pack(const float* __restrict__ kvPart,
                        unsigned* __restrict__ kvpk) {
  const int i = blockIdx.x * blockDim.x + threadIdx.x;   // 32 * 2048
  if (i >= 32 * 2048) return;
  const int bh  = i >> 11;
  const int rem = i & 2047;
  const int m   = rem >> 5;
  const int p   = rem & 31;
  float lo = 0.f, hi = 0.f;
#pragma unroll
  for (int c = 0; c < NCHUNK; ++c) {
    const float* src = kvPart + ((size_t)c * 32 + bh) * (HD * HD);
    lo += src[(2 * p) * HD + m];
    hi += src[(2 * p + 1) * HD + m];
  }
  kvpk[i] = (unsigned)f2bfu(lo) | ((unsigned)f2bfu(hi) << 16);
}

// ------------------- numerator = Q @ kv, normalize by Q.ksum ----------------
// Block = (bh, 64-row slab of N). 128 threads = 4 waves. K = 64 -> 2 steps.
// Q fragments come straight from global (16B-aligned rows); kv tile staged
// into LDS with async b128 copies.
__global__ __launch_bounds__(128) void attn_norm(
    const unsigned short* __restrict__ Qb,
    const unsigned* __restrict__ kvpk,
    const float* __restrict__ ksumPart,
    unsigned short* __restrict__ attn) {
  __shared__ __align__(16) unsigned skv[32 * 64];   // 8 KB, [m][p] pairs
  __shared__ float sksum[64];
  __shared__ float sdenp[128];
  __shared__ float sden[64];

  const int t    = threadIdx.x;
  const int lane = t & 31;
  const int wid  = t >> 5;
  const int g    = lane >> 4;
  const int l16  = lane & 15;
  const int bh   = blockIdx.x;
  const int b    = bh >> 4;
  const int h    = bh & 15;
  const int nb   = blockIdx.y * 64;

  // async-stage kv tile: 128 threads x 64B
  {
    const char* src = (const char*)(kvpk + (size_t)bh * 2048) + t * 64;
    char*       dst = (char*)skv + t * 64;
#pragma unroll
    for (int i = 0; i < 4; ++i) async_b128(src + 16 * i, dst + 16 * i);
  }
  if (t < 64) {
    float s = 0.f;
#pragma unroll
    for (int c = 0; c < NCHUNK; ++c) s += ksumPart[((size_t)c * 32 + bh) * HD + t];
    sksum[t] = s;
  }
  wait_async0();
  __syncthreads();

  // denominator: den[r] = dot(Q[row], ksum); 2 threads per row
  {
    const int r = t >> 1;
    const int half = t & 1;
    const unsigned short* qrow =
        Qb + (size_t)(b * SEQ_N + nb + r) * DIM + h * HD + half * 32;
    float s = 0.f;
#pragma unroll 8
    for (int d = 0; d < 32; ++d) s += bfu2f(qrow[d]) * sksum[half * 32 + d];
    sdenp[t] = s;
  }
  __syncthreads();
  if (t < 64) sden[t] = sdenp[2 * t] + sdenp[2 * t + 1];
  __syncthreads();

  const v8f vzero = {0.f, 0.f, 0.f, 0.f, 0.f, 0.f, 0.f, 0.f};
  v8f acc[4];
#pragma unroll
  for (int mt = 0; mt < 4; ++mt) acc[mt] = vzero;

  const int grow = nb + wid * 16 + l16;
#pragma unroll
  for (int ksStep = 0; ksStep < 2; ++ksStep) {
    union AF { v16bf v; int4 q[2]; } af;
    const unsigned short* src =
        Qb + (size_t)(b * SEQ_N + grow) * DIM + h * HD + ksStep * 32 + g * 16;
    af.q[0] = *(const int4*)src;
    af.q[1] = *(const int4*)(src + 8);
#pragma unroll
    for (int mt = 0; mt < 4; ++mt) {
      // B fragment: 8 contiguous dwords in [m][p] layout
      union BF { v16bf v; int4 q[2]; } bf;
      const unsigned* p = skv + (mt * 16 + l16) * 32 + ksStep * 16 + g * 8;
      bf.q[0] = *(const int4*)(p);
      bf.q[1] = *(const int4*)(p + 4);
      acc[mt] = __builtin_amdgcn_wmma_f32_16x16x32_bf16(
          false, af.v, false, bf.v, (short)0, acc[mt], false, false);
    }
  }

#pragma unroll
  for (int mt = 0; mt < 4; ++mt) {
    const int gn = mt * 16 + l16;
#pragma unroll
    for (int r = 0; r < 8; ++r) {
      const int lr = wid * 16 + r + 8 * g;
      const float den = sden[lr] + 1e-6f;
      const float v = acc[mt][r] / den;
      attn[(size_t)(b * SEQ_N + nb + lr) * DIM + h * HD + gn] = f2bfu(v);
    }
  }
}

// ---------------------------------------------------------------------------
extern "C" void kernel_launch(void* const* d_in, const int* in_sizes, int n_in,
                              void* d_out, int out_size, void* d_ws, size_t ws_size,
                              hipStream_t stream) {
  (void)in_sizes; (void)n_in; (void)out_size; (void)ws_size;

  const float* x  = (const float*)d_in[0];
  const float* Wq = (const float*)d_in[1];
  const float* Wk = (const float*)d_in[2];
  const float* Wv = (const float*)d_in[3];
  const float* Wo = (const float*)d_in[4];
  float* out = (float*)d_out;

  char* ws = (char*)d_ws;
  const size_t SZ_X = (size_t)MROWS * DIM * sizeof(unsigned short);   // 32 MB
  const size_t SZ_W = (size_t)DIM * DIM * sizeof(unsigned short);     // 2 MB

  unsigned short* Xbf = (unsigned short*)(ws);                 // also attn (aliased)
  unsigned short* Wqb = (unsigned short*)(ws + SZ_X);
  unsigned short* Wkb = (unsigned short*)(ws + SZ_X + SZ_W);
  unsigned short* Wvb = (unsigned short*)(ws + SZ_X + 2 * SZ_W);
  unsigned short* Wob = (unsigned short*)(ws + SZ_X + 3 * SZ_W);
  unsigned short* Qb  = (unsigned short*)(ws + SZ_X + 4 * SZ_W);
  unsigned short* Kb  = (unsigned short*)((char*)Qb + SZ_X);
  unsigned short* Vb  = (unsigned short*)((char*)Kb + SZ_X);
  float* kvPart   = (float*)((char*)Vb + SZ_X);                         // 4 MB
  float* ksumPart = (float*)((char*)kvPart + (size_t)NCHUNK * 32 * HD * HD * 4);
  unsigned* kvpk  = (unsigned*)((char*)ksumPart + (size_t)NCHUNK * 32 * HD * 4);
  unsigned short* attn = Xbf;   // X is dead after the QKV GEMMs

  // 1) convert inputs to bf16
  {
    const int nX = MROWS * DIM;
    cvt_f32_to_bf16<<<(nX + 255) / 256, 256, 0, stream>>>(x, Xbf, nX);
    const int nW = DIM * DIM;
    cvt_f32_to_bf16<<<(nW + 255) / 256, 256, 0, stream>>>(Wq, Wqb, nW);
    cvt_f32_to_bf16<<<(nW + 255) / 256, 256, 0, stream>>>(Wk, Wkb, nW);
    cvt_f32_to_bf16<<<(nW + 255) / 256, 256, 0, stream>>>(Wv, Wvb, nW);
    cvt_f32_to_bf16<<<(nW + 255) / 256, 256, 0, stream>>>(Wo, Wob, nW);
  }

  // 2) Q/K/V projections (phi on Q,K)
  dim3 gGemm(DIM / 128, MROWS / 128);
  gemm_xwT<0><<<gGemm, 256, 0, stream>>>(Xbf, Wqb, Qb, nullptr);
  gemm_xwT<0><<<gGemm, 256, 0, stream>>>(Xbf, Wkb, Kb, nullptr);
  gemm_xwT<1><<<gGemm, 256, 0, stream>>>(Xbf, Wvb, Vb, nullptr);

  // 3) kv = K^T V and k_sum, chunked partials then deterministic reduce+pack
  kv_accum<<<dim3(32, NCHUNK), 128, 0, stream>>>(Kb, Vb, kvPart, ksumPart);
  kv_pack<<<(32 * 2048 + 255) / 256, 256, 0, stream>>>(kvPart, kvpk);

  // 4) numerator / denominator / normalize -> attn (bf16, aliases Xbf)
  attn_norm<<<dim3(32, SEQ_N / 64), 128, 0, stream>>>(Qb, kvpk, ksumPart, attn);

  // 5) final projection: out = attn @ Wo^T (f32 output)
  gemm_xwT<2><<<gGemm, 256, 0, stream>>>(attn, Wob, nullptr, out);
}
